// FeedForward_1554778161658
// MI455X (gfx1250) — 8-bit matrix operand order repaired and checked by reading, compiled, not run on hardware
//
#include <hip/hip_runtime.h>
#include <hip/hip_bf16.h>

typedef __attribute__((ext_vector_type(8))) int v8i;

#define EMBED   1024
#define HIDDEN  4096
#define TOKENS  16384
#define EPSQ    1e-5f

__device__ __forceinline__ float gelu_exact(float x) {
    return 0.5f * x * (1.0f + erff(x * 0.70710678118654752440f));
}

__global__ void bitffn_abssum_partial(const float* __restrict__ w,
                                      float* __restrict__ partial, int n4) {
    __shared__ float sm[256];
    float s = 0.f;
    for (int i = blockIdx.x * blockDim.x + threadIdx.x; i < n4;
         i += gridDim.x * blockDim.x) {
        float4 v = ((const float4*)w)[i];
        s += fabsf(v.x) + fabsf(v.y) + fabsf(v.z) + fabsf(v.w);
    }
    sm[threadIdx.x] = s;
    __syncthreads();
    for (int st = 128; st > 0; st >>= 1) {
        if (threadIdx.x < st) sm[threadIdx.x] += sm[threadIdx.x + st];
        __syncthreads();
    }
    if (threadIdx.x == 0) partial[blockIdx.x] = sm[0];
}

__global__ void bitffn_absmean_final(const float* __restrict__ partial,
                                     float* __restrict__ s_out, float invN) {
    __shared__ float sm[256];
    sm[threadIdx.x] = partial[blockIdx.x * 256 + threadIdx.x];
    __syncthreads();
    for (int st = 128; st > 0; st >>= 1) {
        if (threadIdx.x < st) sm[threadIdx.x] += sm[threadIdx.x + st];
        __syncthreads();
    }
    if (threadIdx.x == 0) s_out[blockIdx.x] = fmaxf(sm[0] * invN, EPSQ);
}

__global__ void bitffn_quant_weight(const float* __restrict__ w,
                                    const float* __restrict__ s_ptr,
                                    signed char* __restrict__ wq, int n) {
    const float inv = 1.0f / *s_ptr;
    int i = (blockIdx.x * blockDim.x + threadIdx.x) * 4;
    if (i < n) {
        float4 v = *(const float4*)(w + i);
        char4 q;
        q.x = (signed char)fminf(fmaxf(rintf(v.x * inv), -1.f), 1.f);
        q.y = (signed char)fminf(fmaxf(rintf(v.y * inv), -1.f), 1.f);
        q.z = (signed char)fminf(fmaxf(rintf(v.z * inv), -1.f), 1.f);
        q.w = (signed char)fminf(fmaxf(rintf(v.w * inv), -1.f), 1.f);
        *(char4*)(wq + i) = q;
    }
}

template <int LEN>
__global__ void bitffn_quant_act(const float* __restrict__ x,
                                 signed char* __restrict__ xq,
                                 float* __restrict__ a_inv) {
    __shared__ float sm[256];
    constexpr int PER = LEN / 256;
    const float* row = x + (size_t)blockIdx.x * LEN;
    float vals[PER];
    float m = 0.f;
#pragma unroll
    for (int j = 0; j < PER; ++j) {
        float v = row[threadIdx.x + 256 * j];
        vals[j] = v;
        m = fmaxf(m, fabsf(v));
    }
    sm[threadIdx.x] = m;
    __syncthreads();
    for (int st = 128; st > 0; st >>= 1) {
        if (threadIdx.x < st)
            sm[threadIdx.x] = fmaxf(sm[threadIdx.x], sm[threadIdx.x + st]);
        __syncthreads();
    }
    const float amax  = fmaxf(sm[0], EPSQ);
    const float scale = 127.0f / amax;
    signed char* qrow = xq + (size_t)blockIdx.x * LEN;
#pragma unroll
    for (int j = 0; j < PER; ++j) {
        float q = fminf(fmaxf(rintf(vals[j] * scale), -128.f), 127.f);
        qrow[threadIdx.x + 256 * j] = (signed char)q;
    }
    if (threadIdx.x == 0) a_inv[blockIdx.x] = amax / 127.0f;
}

__device__ __forceinline__ v8i load_afrag(const signed char* arow, int k0) {
    v8i a;
#pragma unroll
    for (int j = 0; j < 4; ++j) {
        int2 v = *(const int2*)(arow + k0 + 32 * (j >> 1) + 8 * (j & 1));
        a[2 * j]     = v.x;
        a[2 * j + 1] = v.y;
    }
    return a;
}
__device__ __forceinline__ v8i load_bfrag(const signed char* brow, int k0) {
    int4 lo = *(const int4*)(brow + k0);
    int4 hi = *(const int4*)(brow + k0 + 32);
    v8i b;
    b[0] = lo.x; b[1] = lo.y; b[2] = lo.z; b[3] = lo.w;
    b[4] = hi.x; b[5] = hi.y; b[6] = hi.z; b[7] = hi.w;
    return b;
}

template <int KDIM, bool DO_GELU>
__global__ __launch_bounds__(256) void bitffn_gemm_iu8(
    const signed char* __restrict__ Aq, const signed char* __restrict__ Bq,
    const float* __restrict__ a_inv, const float* __restrict__ s_w,
    float* __restrict__ Out, int N) {
    const int lane = threadIdx.x & 31;
    const int wave = threadIdx.x >> 5;
    const int half = lane >> 4;
    const int l16  = lane & 15;

    const int m0 = (blockIdx.y * 8 + wave) * 32;
    const int n0 = blockIdx.x * 64;

    const v8i vz = {0, 0, 0, 0, 0, 0, 0, 0};
    v8i acc[8];
#pragma unroll
    for (int i = 0; i < 8; ++i) acc[i] = vz;

    const signed char* arow0 = Aq + (size_t)(m0 + l16) * KDIM + 16 * half;
    const signed char* arow1 = arow0 + (size_t)16 * KDIM;
    const signed char* brow[4];
#pragma unroll
    for (int t = 0; t < 4; ++t)
        brow[t] = Bq + (size_t)(n0 + 16 * t + l16) * KDIM + 16 * half;

    v8i aC0 = load_afrag(arow0, 0);
    v8i aC1 = load_afrag(arow1, 0);
    v8i bC[4];
#pragma unroll
    for (int t = 0; t < 4; ++t) bC[t] = load_bfrag(brow[t], 0);

#pragma unroll 1
    for (int k0 = 0; k0 < KDIM; k0 += 128) {
        v8i aN0 = load_afrag(arow0, k0 + 64);
        v8i aN1 = load_afrag(arow1, k0 + 64);
        v8i bN[4];
#pragma unroll
        for (int t = 0; t < 4; ++t) bN[t] = load_bfrag(brow[t], k0 + 64);

#pragma unroll
        for (int t = 0; t < 4; ++t) {
            acc[t]     = __builtin_amdgcn_wmma_i32_16x16x64_iu8(
                true, aC0, true, bC[t], acc[t], false, false);
            acc[4 + t] = __builtin_amdgcn_wmma_i32_16x16x64_iu8(
                true, aC1, true, bC[t], acc[4 + t], false, false);
        }

        if (k0 + 128 < KDIM) {
            aC0 = load_afrag(arow0, k0 + 128);
            aC1 = load_afrag(arow1, k0 + 128);
#pragma unroll
            for (int t = 0; t < 4; ++t) bC[t] = load_bfrag(brow[t], k0 + 128);
        }

#pragma unroll
        for (int t = 0; t < 4; ++t) {
            acc[t]     = __builtin_amdgcn_wmma_i32_16x16x64_iu8(
                true, aN0, true, bN[t], acc[t], false, false);
            acc[4 + t] = __builtin_amdgcn_wmma_i32_16x16x64_iu8(
                true, aN1, true, bN[t], acc[4 + t], false, false);
        }
    }

    const float sw = *s_w;
#pragma unroll
    for (int mh = 0; mh < 2; ++mh) {
        float rs[8];
#pragma unroll
        for (int r = 0; r < 8; ++r)
            rs[r] = sw * a_inv[m0 + 16 * mh + r + 8 * half];
#pragma unroll
        for (int t = 0; t < 4; ++t) {
#pragma unroll
            for (int r = 0; r < 8; ++r) {
                const int m = m0 + 16 * mh + r + 8 * half;
                const int n = n0 + 16 * t + l16;
                float v = (float)acc[4 * mh + t][r] * rs[r];
                if (DO_GELU) v = gelu_exact(v);
                Out[(size_t)m * N + n] = v;
            }
        }
    }
}

extern "C" void kernel_launch(void* const* d_in, const int* in_sizes, int n_in,
                              void* d_out, int out_size, void* d_ws,
                              size_t ws_size, hipStream_t stream) {
    const float* x  = (const float*)d_in[0];
    const float* w1 = (const float*)d_in[1];
    const float* w2 = (const float*)d_in[2];
    float* out = (float*)d_out;

    const int NW = HIDDEN * EMBED;

    char* ws = (char*)d_ws;
    size_t off = 0;
    signed char* wq1 = (signed char*)(ws + off); off += (size_t)NW;
    signed char* wq2 = (signed char*)(ws + off); off += (size_t)NW;
    signed char* xq  = (signed char*)(ws + off); off += (size_t)TOKENS * EMBED;
    signed char* hq  = (signed char*)(ws + off); off += (size_t)TOKENS * HIDDEN;
    float* h     = (float*)(ws + off); off += (size_t)TOKENS * HIDDEN * 4;
    float* a_inv = (float*)(ws + off); off += (size_t)TOKENS * 4;
    float* h_inv = (float*)(ws + off); off += (size_t)TOKENS * 4;
    float* partials = (float*)(ws + off); off += 512 * 4;
    float* s12      = (float*)(ws + off); off += 2 * 4;

    bitffn_abssum_partial<<<256, 256, 0, stream>>>(w1, partials,       NW / 4);
    bitffn_abssum_partial<<<256, 256, 0, stream>>>(w2, partials + 256, NW / 4);
    bitffn_absmean_final<<<2, 256, 0, stream>>>(partials, s12, 1.0f / (float)NW);

    bitffn_quant_weight<<<NW / 1024, 256, 0, stream>>>(w1, s12 + 0, wq1, NW);
    bitffn_quant_weight<<<NW / 1024, 256, 0, stream>>>(w2, s12 + 1, wq2, NW);

    bitffn_quant_act<EMBED><<<TOKENS, 256, 0, stream>>>(x, xq, a_inv);

    dim3 g1(HIDDEN / 64, TOKENS / 256);
    bitffn_gemm_iu8<EMBED, true><<<g1, 256, 0, stream>>>(
        xq, wq1, a_inv, s12 + 0, h, HIDDEN);

    bitffn_quant_act<HIDDEN><<<TOKENS, 256, 0, stream>>>(h, hq, h_inv);

    dim3 g2(EMBED / 64, TOKENS / 256);
    bitffn_gemm_iu8<HIDDEN, false><<<g2, 256, 0, stream>>>(
        hq, wq2, h_inv, s12 + 1, out, EMBED);
}
